// SelfAttention_42992622633248
// MI455X (gfx1250) — matrix operand order repaired and checked by reading, not compiled, not run on hardware
//
#include <hip/hip_runtime.h>
#include <stdint.h>

typedef __attribute__((ext_vector_type(16))) _Float16 v16h;
typedef __attribute__((ext_vector_type(8)))  _Float16 v8h;
typedef __attribute__((ext_vector_type(8)))  float    v8f;

union V16U { v16h v; struct { v8h lo; v8h hi; } p; };

#define WMMA_F16(a, b, c) \
  __builtin_amdgcn_wmma_f32_16x16x32_f16(false, (a), false, (b), (short)0, (c), false, false)

constexpr int Bn = 16, Cc = 256, Nn = 1024, Gg = 4;
constexpr int GRP_ELEMS = (Cc / Gg) * Nn;   // 65536

// ---------------------------------------------------------------------------
// helpers
// ---------------------------------------------------------------------------

// Load a 16x32 f16 A-fragment. Lane l: row m = l&15, half hs = l>>4.
// a[0..7]  = A[m][kbase + hs*16 + 0..7]
// a[8..15] = A[m][kbase + hs*16 + 8..15]
__device__ __forceinline__ v16h load_a_frag(const _Float16* base, int ld, int kbase, int lane) {
  const int m = lane & 15, hs = lane >> 4;
  const _Float16* p = base + (size_t)m * ld + kbase + hs * 16;
  V16U u;
  u.p.lo = *(const v8h*)(p);
  u.p.hi = *(const v8h*)(p + 8);
  return u.v;
}

// Async global->LDS 128-bit copy (per-lane addresses), tracked by ASYNCcnt.
__device__ __forceinline__ void async_b128(uint32_t lds_off, const void* g) {
  asm volatile("global_load_async_to_lds_b128 %0, %1, off"
               :: "v"(lds_off), "v"((uint64_t)(uintptr_t)g)
               : "memory");
}

// ---------------------------------------------------------------------------
// GroupNorm stats: one block per (batch, group); 65536 contiguous floats.
// ---------------------------------------------------------------------------
__global__ void __launch_bounds__(256) gn_stats_kernel(const float* __restrict__ x,
                                                       float* __restrict__ mean,
                                                       float* __restrict__ rstd) {
  __shared__ float ss[256];
  __shared__ float sq[256];
  const float* p = x + (size_t)blockIdx.x * GRP_ELEMS;
  float s = 0.f, q = 0.f;
  for (int i = threadIdx.x; i < GRP_ELEMS; i += 256) {
    float v = p[i];
    s += v; q += v * v;
  }
  ss[threadIdx.x] = s; sq[threadIdx.x] = q;
  __syncthreads();
  for (int w = 128; w > 0; w >>= 1) {
    if (threadIdx.x < (unsigned)w) {
      ss[threadIdx.x] += ss[threadIdx.x + w];
      sq[threadIdx.x] += sq[threadIdx.x + w];
    }
    __syncthreads();
  }
  if (threadIdx.x == 0) {
    float mu  = ss[0] / (float)GRP_ELEMS;
    float var = sq[0] / (float)GRP_ELEMS - mu * mu;
    mean[blockIdx.x] = mu;
    rstd[blockIdx.x] = rsqrtf(var + 1e-5f);
  }
}

// ---------------------------------------------------------------------------
// GroupNorm apply + transpose [B,C,N] -> h16[b][n][c] (f16)
// ---------------------------------------------------------------------------
__global__ void __launch_bounds__(256) gn_apply_kernel(const float* __restrict__ x,
                                                       const float* __restrict__ gamma,
                                                       const float* __restrict__ beta,
                                                       const float* __restrict__ mean,
                                                       const float* __restrict__ rstd,
                                                       _Float16* __restrict__ h16) {
  int t  = blockIdx.x * 256 + threadIdx.x;  // 0 .. B*C*(N/4)-1
  int nq = t & 255;
  int c  = (t >> 8) & 255;
  int b  = t >> 16;
  int g  = c >> 6;
  float mu = mean[b * 4 + g];
  float rs = rstd[b * 4 + g];
  float ga = gamma[c] * rs;
  float be = beta[c] - mu * ga;
  const float4 xv = *(const float4*)(x + (((size_t)b * Cc + c) << 10) + nq * 4);
  size_t base = ((size_t)b * Nn + nq * 4) * Cc + c;
  h16[base          ] = (_Float16)(xv.x * ga + be);
  h16[base + Cc     ] = (_Float16)(xv.y * ga + be);
  h16[base + 2 * Cc ] = (_Float16)(xv.z * ga + be);
  h16[base + 3 * Cc ] = (_Float16)(xv.w * ga + be);
}

// ---------------------------------------------------------------------------
// Cast the three 256x256 weights to f16 (row-major [d][c], unchanged).
// ---------------------------------------------------------------------------
__global__ void __launch_bounds__(256) cast_w_kernel(const float* __restrict__ w0,
                                                     const float* __restrict__ w1,
                                                     const float* __restrict__ w2,
                                                     _Float16* __restrict__ dst) {
  int i = blockIdx.x * 256 + threadIdx.x;          // 0..65535
  const float* src = (blockIdx.y == 0) ? w0 : (blockIdx.y == 1 ? w1 : w2);
  dst[(size_t)blockIdx.y * 65536 + i] = (_Float16)src[i];
}

// ---------------------------------------------------------------------------
// QKV projection GEMM: y[m][d] = sum_c h[m][c] * W[d][c] + bias[d]
// grid = (M/64, C/64); block = 128 (4 waves, 16 rows each, 64 d-cols).
// If out_tn != null: store transposed f16 [b][d][n] (for V). Else store
// f16 [m][d] scaled by `scale` (1/16 for Q).
// ---------------------------------------------------------------------------
__global__ void __launch_bounds__(128) qkv_kernel(const _Float16* __restrict__ h16,
                                                  const _Float16* __restrict__ w16,
                                                  const float* __restrict__ bias,
                                                  _Float16* __restrict__ out_nd,
                                                  _Float16* __restrict__ out_tn,
                                                  float scale) {
  const int lane = threadIdx.x & 31;
  const int wave = threadIdx.x >> 5;
  const int mrow0 = blockIdx.x * 64 + wave * 16;
  const int d0 = blockIdx.y * 64;
  const int n15 = lane & 15, hs = lane >> 4;

  v8f acc[4] = {};
  const _Float16* hrow = h16 + (size_t)mrow0 * Cc;

#pragma unroll
  for (int k = 0; k < 8; ++k) {
    v16h a = load_a_frag(hrow, Cc, k * 32, lane);
    const int kb = k * 32 + hs * 16;
#pragma unroll
    for (int dt = 0; dt < 4; ++dt) {
      int d = d0 + dt * 16 + n15;
      v16h b = *(const v16h*)(w16 + (size_t)d * Cc + kb);   // B[k][n] = W[d][k]
      acc[dt] = WMMA_F16(a, b, acc[dt]);
    }
  }

  if (out_tn) {
    const int bb = mrow0 >> 10;
    const int nloc = (mrow0 & 1023) + hs * 8;
#pragma unroll
    for (int dt = 0; dt < 4; ++dt) {
      int d = d0 + dt * 16 + n15;
      float bv = bias[d];
      v8h pk;
#pragma unroll
      for (int r = 0; r < 8; ++r) pk[r] = (_Float16)(acc[dt][r] + bv);
      *(v8h*)(out_tn + ((size_t)bb * Cc + d) * Nn + nloc) = pk;
    }
  } else {
#pragma unroll
    for (int dt = 0; dt < 4; ++dt) {
      int d = d0 + dt * 16 + n15;
      float bv = bias[d];
#pragma unroll
      for (int r = 0; r < 8; ++r) {
        int m = mrow0 + hs * 8 + r;
        out_nd[(size_t)m * Cc + d] = (_Float16)((acc[dt][r] + bv) * scale);
      }
    }
  }
}

// ---------------------------------------------------------------------------
// Flash attention: grid = (N/64, B); block = 128 (4 waves x 16 q-rows).
// K/V^T chunks (32 keys) staged to LDS with async b128 copies.
// Epilogue adds residual x and writes fp32 [B,C,H,W].
// ---------------------------------------------------------------------------
__global__ void __launch_bounds__(128) attn_kernel(const _Float16* __restrict__ q16,
                                                   const _Float16* __restrict__ k16,
                                                   const _Float16* __restrict__ vt16,
                                                   const float* __restrict__ x,
                                                   float* __restrict__ out) {
  __shared__ __align__(128) _Float16 Klds[32 * 256];     // [key][d]  16 KB
  __shared__ __align__(128) _Float16 VTlds[256 * 32];    // [d][key]  16 KB
  __shared__ __align__(128) _Float16 Plds[4][16 * 32];   // per-wave P  4 KB

  const int lane = threadIdx.x & 31;
  const int wave = threadIdx.x >> 5;
  const int b = blockIdx.y;
  const int n0w = blockIdx.x * 64 + wave * 16;   // q-row base of this wave
  const int n15 = lane & 15, hs = lane >> 4;

  // Q fragments, kept in registers (q already scaled by 1/sqrt(C))
  v16h qf[8];
  const _Float16* qrow = q16 + ((size_t)b * Nn + n0w) * Cc;
#pragma unroll
  for (int k = 0; k < 8; ++k) qf[k] = load_a_frag(qrow, Cc, k * 32, lane);

  v8f O[16] = {};
  float mrow[8], lrow[8];
#pragma unroll
  for (int r = 0; r < 8; ++r) { mrow[r] = -1e30f; lrow[r] = 0.f; }

  const uint32_t kldsOff = (uint32_t)(uintptr_t)(&Klds[0]);
  const uint32_t vldsOff = (uint32_t)(uintptr_t)(&VTlds[0]);
  const _Float16* kbase  = k16  + (size_t)b * Nn * Cc;
  const _Float16* vtbase = vt16 + (size_t)b * Cc * Nn;
  _Float16* myP = &Plds[wave][0];

  for (int j0 = 0; j0 < Nn; j0 += 32) {
    __syncthreads();   // previous chunk's LDS consumers are done

    // --- stage K chunk: 32 rows x 512B, contiguous 16KB
    {
      const char* src = (const char*)(kbase + (size_t)j0 * Cc);
#pragma unroll
      for (int i = 0; i < 4; ++i) {
        uint32_t off = (threadIdx.x + i * 128) * 32u;      // 2 x b128 per iter
        async_b128(kldsOff + off,      src + off);
        async_b128(kldsOff + off + 16, src + off + 16);
      }
      // --- stage V^T chunk: 256 rows of 64B, gathered (stride 2KB in global)
#pragma unroll
      for (int i = 0; i < 8; ++i) {
        uint32_t qb = threadIdx.x + i * 128;               // 16B-unit index 0..1023
        uint32_t d = qb >> 2, part = qb & 3;
        const char* s = (const char*)(vtbase + (size_t)d * Nn + j0) + part * 16;
        async_b128(vldsOff + qb * 16u, s);
      }
      asm volatile("s_wait_asynccnt 0x0" ::: "memory");
    }
    __syncthreads();   // all waves' staging visible

    // --- S = Q @ K^T : two 16x16 tiles over 32 keys
    v8f s0 = {}, s1 = {};
#pragma unroll
    for (int k = 0; k < 8; ++k) {
      const int kb = k * 32 + hs * 16;
      v16h b0 = *(const v16h*)(&Klds[(size_t)(n15)      * Cc + kb]);
      v16h b1 = *(const v16h*)(&Klds[(size_t)(16 + n15) * Cc + kb]);
      s0 = WMMA_F16(qf[k], b0, s0);
      s1 = WMMA_F16(qf[k], b1, s1);
    }

    // --- online softmax update; rows live in one VGPR across a 16-lane half
#pragma unroll
    for (int r = 0; r < 8; ++r) {
      float v0 = s0[r], v1 = s1[r];
      float mx = fmaxf(v0, v1);
      mx = fmaxf(mx, __shfl_xor(mx, 1));
      mx = fmaxf(mx, __shfl_xor(mx, 2));
      mx = fmaxf(mx, __shfl_xor(mx, 4));
      mx = fmaxf(mx, __shfl_xor(mx, 8));
      float mnew  = fmaxf(mrow[r], mx);
      float alpha = __expf(mrow[r] - mnew);
      float p0 = __expf(v0 - mnew);
      float p1 = __expf(v1 - mnew);
      float rs = p0 + p1;
      rs += __shfl_xor(rs, 1);
      rs += __shfl_xor(rs, 2);
      rs += __shfl_xor(rs, 4);
      rs += __shfl_xor(rs, 8);
      lrow[r] = lrow[r] * alpha + rs;
      mrow[r] = mnew;
#pragma unroll
      for (int dn = 0; dn < 16; ++dn) O[dn][r] *= alpha;
      int m = hs * 8 + r;
      myP[m * 32 + n15]      = (_Float16)p0;
      myP[m * 32 + 16 + n15] = (_Float16)p1;
    }
    // intra-wave LDS RAW: stores above -> A-fragment load below
    asm volatile("s_wait_dscnt 0x0" ::: "memory");

    // --- O += P @ V (B-fragments from transposed V: contiguous 32B in LDS)
    v16h pf = load_a_frag(myP, 32, 0, lane);
#pragma unroll
    for (int dn = 0; dn < 16; ++dn) {
      v16h bf = *(const v16h*)(&VTlds[(size_t)(dn * 16 + n15) * 32 + hs * 16]);
      O[dn] = WMMA_F16(pf, bf, O[dn]);
    }
  }

  // --- epilogue: out[b][c][n] = O/l + x  (8 contiguous floats per lane)
#pragma unroll
  for (int dn = 0; dn < 16; ++dn) {
    int d = dn * 16 + n15;
    size_t base = ((size_t)b * Cc + d) * Nn + n0w + hs * 8;
    const float* xr = x + base;
    float* orow = out + base;
    float inv_l[1];
#pragma unroll
    for (int r = 0; r < 8; ++r) {
      orow[r] = O[dn][r] / lrow[r] + xr[r];
    }
    (void)inv_l;
  }
}

// ---------------------------------------------------------------------------
// launch
// ---------------------------------------------------------------------------
extern "C" void kernel_launch(void* const* d_in, const int* in_sizes, int n_in,
                              void* d_out, int out_size, void* d_ws, size_t ws_size,
                              hipStream_t stream) {
  const float* x     = (const float*)d_in[0];
  const float* Wq    = (const float*)d_in[1];
  const float* bq    = (const float*)d_in[2];
  const float* Wk    = (const float*)d_in[3];
  const float* bk    = (const float*)d_in[4];
  const float* Wv    = (const float*)d_in[5];
  const float* bv    = (const float*)d_in[6];
  const float* gamma = (const float*)d_in[7];
  const float* beta  = (const float*)d_in[8];
  float* out = (float*)d_out;

  char* wsb = (char*)d_ws;
  float*    mean = (float*)wsb;                         // 64 f32
  float*    rstd = (float*)(wsb + 256);                 // 64 f32
  _Float16* h16  = (_Float16*)(wsb + 1024);             // [B][N][C]  8 MB
  _Float16* w16  = h16 + (size_t)Bn * Nn * Cc;          // 3 x 256x256 f16
  _Float16* q16  = w16 + 3 * Cc * Cc;                   // [B][N][C]  8 MB
  _Float16* k16  = q16 + (size_t)Bn * Nn * Cc;          // [B][N][C]  8 MB
  _Float16* vt16 = k16 + (size_t)Bn * Nn * Cc;          // [B][C][N]  8 MB

  gn_stats_kernel<<<Bn * Gg, 256, 0, stream>>>(x, mean, rstd);
  gn_apply_kernel<<<(Bn * Cc * (Nn / 4)) / 256, 256, 0, stream>>>(x, gamma, beta, mean, rstd, h16);
  cast_w_kernel<<<dim3(256, 3), 256, 0, stream>>>(Wq, Wk, Wv, w16);

  const int Mtiles = (Bn * Nn) / 64;   // 256
  qkv_kernel<<<dim3(Mtiles, 4), 128, 0, stream>>>(h16, w16,                 bq, q16, nullptr, 0.0625f);
  qkv_kernel<<<dim3(Mtiles, 4), 128, 0, stream>>>(h16, w16 + Cc * Cc,       bk, k16, nullptr, 1.0f);
  qkv_kernel<<<dim3(Mtiles, 4), 128, 0, stream>>>(h16, w16 + 2 * Cc * Cc,   bv, nullptr, vt16, 1.0f);

  attn_kernel<<<dim3(Nn / 64, Bn), 128, 0, stream>>>(q16, k16, vt16, x, out);
}
